// GPPT_4020089389577
// MI455X (gfx1250) — matrix operand order repaired and checked by reading, not compiled, not run on hardware
//
#include <hip/hip_runtime.h>

// ---------------- problem constants (fixed by the reference) ----------------
#define NN   20000      // nodes
#define EE   320000     // edges
#define HH   256        // hidden / in_feats
#define CC   64         // classes
#define NH   ((size_t)NN * HH)

typedef __attribute__((ext_vector_type(16))) __bf16 v16bf;
typedef __attribute__((ext_vector_type(8)))  __bf16 v8bf;
typedef __attribute__((ext_vector_type(8)))  float  v8f;

// ---------------------------------------------------------------------------
// Small utility kernels
// ---------------------------------------------------------------------------
__global__ void k_zero4(float4* p, int n4) {
    int i = blockIdx.x * blockDim.x + threadIdx.x;
    if (i < n4) p[i] = make_float4(0.f, 0.f, 0.f, 0.f);
}

__global__ void k_cvt_bf16(const float* __restrict__ in, __bf16* __restrict__ out, int n) {
    int i = blockIdx.x * blockDim.x + threadIdx.x;
    if (i < n) out[i] = (__bf16)in[i];
}

// agg = msg / max(deg,1), converted to bf16
__global__ void k_div_cvt(const float* __restrict__ msg, const float* __restrict__ deg,
                          __bf16* __restrict__ out) {
    int i = blockIdx.x * blockDim.x + threadIdx.x;
    if (i < (int)NH) {
        float d = deg[i >> 8];
        out[i] = (__bf16)(msg[i] / fmaxf(d, 1.0f));
    }
}

// ---------------------------------------------------------------------------
// Edge scatter: one wave per edge; lane handles 8 contiguous features.
// feat[src] is L2-resident (20 MB << 192 MB L2) -> atomics hit L2.
// ---------------------------------------------------------------------------
__global__ void __launch_bounds__(256)
k_scatter(const float* __restrict__ feat, const long long* __restrict__ ei,
          float* __restrict__ msg, float* __restrict__ deg, int addDeg) {
    int gw   = (blockIdx.x * blockDim.x + threadIdx.x) >> 5;
    int lane = threadIdx.x & 31;
    if (gw >= EE) return;
    long long s = ei[gw];
    long long d = ei[EE + gw];
    const float4* f4 = (const float4*)(feat + (size_t)s * HH) + lane * 2;
    float4 a = f4[0], b = f4[1];
    float* md = msg + (size_t)d * HH + lane * 8;
    __hip_atomic_fetch_add(md + 0, a.x, __ATOMIC_RELAXED, __HIP_MEMORY_SCOPE_AGENT);
    __hip_atomic_fetch_add(md + 1, a.y, __ATOMIC_RELAXED, __HIP_MEMORY_SCOPE_AGENT);
    __hip_atomic_fetch_add(md + 2, a.z, __ATOMIC_RELAXED, __HIP_MEMORY_SCOPE_AGENT);
    __hip_atomic_fetch_add(md + 3, a.w, __ATOMIC_RELAXED, __HIP_MEMORY_SCOPE_AGENT);
    __hip_atomic_fetch_add(md + 4, b.x, __ATOMIC_RELAXED, __HIP_MEMORY_SCOPE_AGENT);
    __hip_atomic_fetch_add(md + 5, b.y, __ATOMIC_RELAXED, __HIP_MEMORY_SCOPE_AGENT);
    __hip_atomic_fetch_add(md + 6, b.z, __ATOMIC_RELAXED, __HIP_MEMORY_SCOPE_AGENT);
    __hip_atomic_fetch_add(md + 7, b.w, __ATOMIC_RELAXED, __HIP_MEMORY_SCOPE_AGENT);
    if (addDeg && lane == 0)
        __hip_atomic_fetch_add(deg + d, 1.0f, __ATOMIC_RELAXED, __HIP_MEMORY_SCOPE_AGENT);
}

// ---------------------------------------------------------------------------
// WMMA bf16 GEMM:  out[m,o] = relu( A1@W1^T + A2@W2^T + bias )
// A*: [N,256] bf16 row-major.  W*: [256,256] bf16 row-major ([out,in]).
// One wave computes a 16(M) x 64(N) strip. K=256 in steps of 32.
//
// Fragment addressing used here (wave32), contiguous k order, same for A and B:
//   A 16x32: lane L (m=L&15, h=L>>4) holds row m, K in {16h..16h+15}
//            -> two 16B contiguous loads from the activation row.
//   B 32x16: lane L (n=L&15, h=L>>4) holds col n, K in {16h..16h+15}
//            -> since B = W^T, B[k][n] = W[n][k]: one 32B contiguous load
//               from weight row n.
// ---------------------------------------------------------------------------
__device__ __forceinline__ v16bf load_a_frag(const __bf16* p) {
    v8bf lo = *(const v8bf*)p;
    v8bf hi = *(const v8bf*)(p + 8);
    v16bf r;
#pragma unroll
    for (int i = 0; i < 8; i++) { r[i] = lo[i]; r[i + 8] = hi[i]; }
    return r;
}

__device__ __forceinline__ void gemm_quarter(const __bf16* __restrict__ A,
                                             const __bf16* __restrict__ W,
                                             int m0, int o0, int n, int h, v8f acc[4]) {
    const __bf16* ar = A + (size_t)(m0 + n) * HH + h * 16;
    const __bf16* w0 = W + (size_t)(o0 +  0 + n) * HH + h * 16;
    const __bf16* w1 = W + (size_t)(o0 + 16 + n) * HH + h * 16;
    const __bf16* w2 = W + (size_t)(o0 + 32 + n) * HH + h * 16;
    const __bf16* w3 = W + (size_t)(o0 + 48 + n) * HH + h * 16;
#pragma unroll
    for (int k0 = 0; k0 < HH; k0 += 32) {
        v16bf a = load_a_frag(ar + k0);
        v16bf b0 = *(const v16bf*)(w0 + k0);
        v16bf b1 = *(const v16bf*)(w1 + k0);
        v16bf b2 = *(const v16bf*)(w2 + k0);
        v16bf b3 = *(const v16bf*)(w3 + k0);
        acc[0] = __builtin_amdgcn_wmma_f32_16x16x32_bf16(false, a, false, b0, (short)0, acc[0], false, false);
        acc[1] = __builtin_amdgcn_wmma_f32_16x16x32_bf16(false, a, false, b1, (short)0, acc[1], false, false);
        acc[2] = __builtin_amdgcn_wmma_f32_16x16x32_bf16(false, a, false, b2, (short)0, acc[2], false, false);
        acc[3] = __builtin_amdgcn_wmma_f32_16x16x32_bf16(false, a, false, b3, (short)0, acc[3], false, false);
    }
}

__global__ void __launch_bounds__(128)
k_gemm_sage(const __bf16* __restrict__ A1, const __bf16* __restrict__ W1,
            const __bf16* __restrict__ A2, const __bf16* __restrict__ W2,
            const float*  __restrict__ bias,
            float* __restrict__ outF, __bf16* __restrict__ outBf, int writeBf) {
    int wave = threadIdx.x >> 5;       // 0..3 -> 64-wide column strip
    int lane = threadIdx.x & 31;
    int n = lane & 15, h = lane >> 4;
    int m0 = blockIdx.x * 16;          // 1250 row tiles (20000 = 16*1250 exactly)
    int o0 = wave * 64;

    v8f acc[4] = {};
    gemm_quarter(A1, W1, m0, o0, n, h, acc);   // lin_l(agg)
    gemm_quarter(A2, W2, m0, o0, n, h, acc);   // + lin_r(x)

#pragma unroll
    for (int j = 0; j < 4; j++) {
        int col = o0 + j * 16 + n;
        float bv = bias[col];
#pragma unroll
        for (int r = 0; r < 8; r++) {
            int row = m0 + h * 8 + r;               // C/D layout: VGPR r -> M = r + 8h
            float v = fmaxf(acc[j][r] + bv, 0.0f);  // relu
            outF[(size_t)row * HH + col] = v;
            if (writeBf) outBf[(size_t)row * HH + col] = (__bf16)v;
        }
    }
}

// ---------------------------------------------------------------------------
// Routed head: one wave per node. Stage h2 row in LDS, K=3 routing dots with
// wave shfl-xor reduction, first-max argmax, then 2 output channels per lane
// against the selected Wpp[idx] head (contiguous float4 streams).
// ---------------------------------------------------------------------------
__global__ void __launch_bounds__(256)
k_head(const float* __restrict__ h, const float* __restrict__ Wp,
       const float* __restrict__ Wpp, float* __restrict__ out) {
    __shared__ float sh[8 * HH];
    int wave = threadIdx.x >> 5, lane = threadIdx.x & 31;
    int node = blockIdx.x * 8 + wave;              // 2500 blocks * 8 = 20000 exactly

    float* srow = sh + wave * HH;
    const float4* hrow = (const float4*)(h + (size_t)node * HH);
    float4 v0 = hrow[lane * 2], v1 = hrow[lane * 2 + 1];
    ((float4*)srow)[lane * 2]     = v0;
    ((float4*)srow)[lane * 2 + 1] = v1;

    float p[3];
#pragma unroll
    for (int k = 0; k < 3; k++) {
        const float* w = Wp + k * HH + lane * 8;
        float s = v0.x * w[0] + v0.y * w[1] + v0.z * w[2] + v0.w * w[3]
                + v1.x * w[4] + v1.y * w[5] + v1.z * w[6] + v1.w * w[7];
#pragma unroll
        for (int m = 16; m >= 1; m >>= 1) s += __shfl_xor(s, m, 32);
        p[k] = s;
    }
    int idx = 0;
    if (p[1] > p[0])   idx = 1;    // strict '>' == jnp.argmax first-max semantics
    if (p[2] > p[idx]) idx = 2;

    __syncthreads();

#pragma unroll
    for (int t = 0; t < 2; t++) {
        int c = lane + t * 32;
        const float4* w  = (const float4*)(Wpp + ((size_t)idx * CC + c) * HH);
        const float4* s4 = (const float4*)srow;
        float acc = 0.f;
#pragma unroll 4
        for (int d = 0; d < HH / 4; d++) {
            float4 a = s4[d], b = w[d];
            acc += a.x * b.x + a.y * b.y + a.z * b.z + a.w * b.w;
        }
        out[(size_t)node * CC + c] = acc;
    }
}

// ---------------------------------------------------------------------------
static inline size_t take(size_t& cur, size_t bytes) {
    size_t o = cur;
    cur += (bytes + 511) & ~(size_t)511;
    return o;
}

extern "C" void kernel_launch(void* const* d_in, const int* in_sizes, int n_in,
                              void* d_out, int out_size, void* d_ws, size_t ws_size,
                              hipStream_t stream) {
    (void)in_sizes; (void)n_in; (void)out_size; (void)ws_size;

    const float*     x   = (const float*)d_in[0];
    const long long* ei  = (const long long*)d_in[1];   // int64 edge_index [2,E]
    const float*     Wl1 = (const float*)d_in[2];
    const float*     bl1 = (const float*)d_in[3];
    const float*     Wr1 = (const float*)d_in[4];
    const float*     Wl2 = (const float*)d_in[5];
    const float*     bl2 = (const float*)d_in[6];
    const float*     Wr2 = (const float*)d_in[7];
    const float*     Wp  = (const float*)d_in[8];
    const float*     Wpp = (const float*)d_in[9];
    float*           out = (float*)d_out;

    char* ws = (char*)d_ws;
    size_t cur = 0;
    float*  deg   = (float*)(ws + take(cur, (size_t)NN * sizeof(float)));
    float*  msg   = (float*)(ws + take(cur, NH * sizeof(float)));
    float*  h1f   = (float*)(ws + take(cur, NH * sizeof(float)));
    float*  h2f   = (float*)(ws + take(cur, NH * sizeof(float)));
    __bf16* xbf   = (__bf16*)(ws + take(cur, NH * sizeof(__bf16)));
    __bf16* h1bf  = (__bf16*)(ws + take(cur, NH * sizeof(__bf16)));
    __bf16* aggbf = (__bf16*)(ws + take(cur, NH * sizeof(__bf16)));
    __bf16* wl1bf = (__bf16*)(ws + take(cur, (size_t)HH * HH * sizeof(__bf16)));
    __bf16* wr1bf = (__bf16*)(ws + take(cur, (size_t)HH * HH * sizeof(__bf16)));
    __bf16* wl2bf = (__bf16*)(ws + take(cur, (size_t)HH * HH * sizeof(__bf16)));
    __bf16* wr2bf = (__bf16*)(ws + take(cur, (size_t)HH * HH * sizeof(__bf16)));

    const int T = 256;
    // --- prep: zero accumulators, convert x + weights to bf16 ---
    k_zero4<<<dim3((NN / 4 + T - 1) / T), dim3(T), 0, stream>>>((float4*)deg, NN / 4);
    k_zero4<<<dim3((int)(NH / 4 / T)), dim3(T), 0, stream>>>((float4*)msg, (int)(NH / 4));
    k_cvt_bf16<<<dim3((int)(NH / T)), dim3(T), 0, stream>>>(x, xbf, (int)NH);
    k_cvt_bf16<<<dim3(HH * HH / T), dim3(T), 0, stream>>>(Wl1, wl1bf, HH * HH);
    k_cvt_bf16<<<dim3(HH * HH / T), dim3(T), 0, stream>>>(Wr1, wr1bf, HH * HH);
    k_cvt_bf16<<<dim3(HH * HH / T), dim3(T), 0, stream>>>(Wl2, wl2bf, HH * HH);
    k_cvt_bf16<<<dim3(HH * HH / T), dim3(T), 0, stream>>>(Wr2, wr2bf, HH * HH);

    // --- layer 1: scatter-mean(x) -> WMMA GEMM + relu ---
    k_scatter<<<dim3(EE * 32 / T), dim3(T), 0, stream>>>(x, ei, msg, deg, 1);
    k_div_cvt<<<dim3((int)(NH / T)), dim3(T), 0, stream>>>(msg, deg, aggbf);
    k_gemm_sage<<<dim3(NN / 16), dim3(128), 0, stream>>>(aggbf, wl1bf, xbf, wr1bf,
                                                         bl1, h1f, h1bf, 1);

    // --- layer 2: scatter-mean(h1) -> WMMA GEMM + relu (degree reused) ---
    k_zero4<<<dim3((int)(NH / 4 / T)), dim3(T), 0, stream>>>((float4*)msg, (int)(NH / 4));
    k_scatter<<<dim3(EE * 32 / T), dim3(T), 0, stream>>>(h1f, ei, msg, nullptr, 0);
    k_div_cvt<<<dim3((int)(NH / T)), dim3(T), 0, stream>>>(msg, deg, aggbf);
    k_gemm_sage<<<dim3(NN / 16), dim3(128), 0, stream>>>(aggbf, wl2bf, h1bf, wr2bf,
                                                         bl2, h2f, nullptr, 0);

    // --- routed per-center head ---
    k_head<<<dim3(NN / 8), dim3(T), 0, stream>>>(h2f, Wp, Wpp, out);
}
